// SwinTransformer_14860586844292
// MI455X (gfx1250) — matrix operand order repaired and checked by reading, not compiled, not run on hardware
//
#include <hip/hip_runtime.h>
#include <hip/hip_bf16.h>
#include <stdint.h>

// ---------------------------------------------------------------------------
// Swin Transformer (2 blocks) for gfx1250 — f16 WMMA (16x16x32) + TDM staging.
// ---------------------------------------------------------------------------

typedef __attribute__((ext_vector_type(16))) _Float16 v16h;
typedef __attribute__((ext_vector_type(8)))  float    v8f;
typedef __attribute__((ext_vector_type(4)))  unsigned v4u;
typedef __attribute__((ext_vector_type(8)))  int      v8i;
typedef __attribute__((ext_vector_type(4)))  int      v4i;

#define TOKENS  65536   // B*H*W = 16*64*64
#define CDIM    192
#define NHEADS  6
#define HDIM    32
#define NWIN    64      // (64/8)^2 windows per image
#define NTOK    64      // tokens per window
#define QKVDIM  576
#define FFDIM   768
#define TBLSZ   225

__device__ __forceinline__ v8f wmma_f16(v16h a, v16h b, v8f c) {
    return __builtin_amdgcn_wmma_f32_16x16x32_f16(
        /*neg_a=*/false, a, /*neg_b=*/false, b,
        /*c_mod=*/(short)0, c, /*reuse_a=*/false, /*reuse_b=*/false);
}

// ISA 7.12.2: 16-bit A (16x32). lane&15 = M row; element e -> K index.
__device__ __forceinline__ int kmapA(int e, int half) {
    return ((e >> 3) << 4) | (half << 3) | (e & 7);
}
// 16-bit B (32x16). lane&15 = N col; element e -> K index.
__device__ __forceinline__ int kmapB(int e, int half) {
    return ((e >> 3) << 4) | (half << 3) | (e & 7);
}

// ---------------------------------------------------------------------------
// TDM: load a 64(row) x 32(half) tile, row stride 576 halves, into LDS with
// hardware padding of 4 DWORDs every 16 DWORDs -> [64][40]-half layout.
// D# group layouts per CDNA5 ISA.
// ---------------------------------------------------------------------------
__device__ __forceinline__ void tdm_load_qkv_tile(unsigned lds_off, const void* gaddr) {
    unsigned long long ga = (unsigned long long)(uintptr_t)gaddr;
    v4u g0;
    g0[0] = 1u;                                   // count=1, user descriptor
    g0[1] = lds_off;                              // lds_addr (bytes)
    g0[2] = (unsigned)(ga & 0xFFFFFFFFu);         // global_addr[31:0]
    g0[3] = ((unsigned)(ga >> 32) & 0x01FFFFFFu)  // global_addr[56:32]
            | (2u << 30);                         // type = 2 ("image")
    v8i g1;
    g1[0] = (1 << 16)      // data_size = 2 bytes
          | (1 << 20)      // pad_enable
          | (3 << 22)      // pad_interval: 16 DWORDs (one 32-half row)
          | (3 << 25);     // pad_amount: 4 DWORDs (8 halves)
    g1[1] = (32 << 16);    // tensor_dim0 = 32 (bits 79:48, low half)
    g1[2] = (64 << 16);    // tensor_dim1 = 64 (bits 111:80, low half)
    g1[3] = (32 << 16);    // tile_dim0  = 32 (bits 127:112)
    g1[4] = 64;            // tile_dim1  = 64 (bits 143:128)
    g1[5] = QKVDIM;        // tensor_dim0_stride = 576 (bits 207:160, low)
    g1[6] = 0;
    g1[7] = 0;
    v4i z = {0, 0, 0, 0};
#if __clang_major__ >= 23
    v8i z8 = {0, 0, 0, 0, 0, 0, 0, 0};
    __builtin_amdgcn_tensor_load_to_lds(g0, g1, z, z, z8, 0);
#else
    __builtin_amdgcn_tensor_load_to_lds(g0, g1, z, z, 0);
#endif
}

// ---------------------------------------------------------------------------
// Weight f32 -> f16 conversion
// ---------------------------------------------------------------------------
__global__ void k_convert_f16(const float* __restrict__ src,
                              _Float16* __restrict__ dst, int n) {
    int i = blockIdx.x * blockDim.x + threadIdx.x;
    if (i < n) dst[i] = (_Float16)src[i];
}

// ---------------------------------------------------------------------------
// Relative-position bias + shifted-window mask, materialized per depth:
// bm[head][win][i][j]  (6 * 64 * 64 * 64 f32 = 6.3 MB)
// ---------------------------------------------------------------------------
__device__ __forceinline__ int swin_region(int g) {
    return (g < 56) ? 0 : ((g < 60) ? 1 : 2);
}

__global__ void k_biasmask(const float* __restrict__ tbl,
                           float* __restrict__ bm, int shift) {
    int blk  = blockIdx.x;           // head*64 + win
    int head = blk >> 6, win = blk & 63;
    int wh = win >> 3, ww = win & 7;
    for (int p = threadIdx.x; p < NTOK * NTOK; p += blockDim.x) {
        int i = p >> 6, j = p & 63;
        int ri = i >> 3, ci = i & 7, rj = j >> 3, cj = j & 7;
        int idx = (ri - rj + 7) * 15 + (ci - cj + 7);
        float v = tbl[idx * NHEADS + head];
        if (shift) {
            int idi = swin_region(wh * 8 + ri) * 3 + swin_region(ww * 8 + ci);
            int idj = swin_region(wh * 8 + rj) * 3 + swin_region(ww * 8 + cj);
            if (idi != idj) v -= 100.0f;
        }
        bm[(size_t)blk * (NTOK * NTOK) + p] = v;
    }
}

// ---------------------------------------------------------------------------
// LayerNorm (+ optional roll-shift + window partition), f32 in -> f16 out.
// One wave32 per token; blockDim = 256 (8 tokens).
// ---------------------------------------------------------------------------
__global__ void k_layernorm(const float* __restrict__ x,
                            const float* __restrict__ g,
                            const float* __restrict__ b,
                            _Float16* __restrict__ out,
                            int shift, int winpart) {
    int lane  = threadIdx.x & 31;
    int wv    = threadIdx.x >> 5;
    int token = blockIdx.x * 8 + wv;

    const float* row = x + (size_t)token * CDIM;
    float v[6], s = 0.f, ss = 0.f;
#pragma unroll
    for (int j = 0; j < 6; ++j) {
        v[j] = row[lane + 32 * j];
        s += v[j]; ss += v[j] * v[j];
    }
#pragma unroll
    for (int off = 16; off > 0; off >>= 1) {
        s  += __shfl_xor(s,  off, 32);
        ss += __shfl_xor(ss, off, 32);
    }
    const float mu  = s * (1.0f / CDIM);
    const float var = ss * (1.0f / CDIM) - mu * mu;
    const float inv = rsqrtf(var + 1e-5f);

    size_t orow;
    if (winpart) {
        int bb  = token >> 12;
        int pos = token & 4095;
        int h = pos >> 6, w = pos & 63;
        int gh = (h - shift + 64) & 63;   // roll(-shift)
        int gw = (w - shift + 64) & 63;
        int win = ((gh >> 3) << 3) + (gw >> 3);
        int n   = ((gh & 7) << 3) + (gw & 7);
        orow = (size_t)bb * 4096 + win * NTOK + n;
    } else {
        orow = token;
    }
    _Float16* o = out + orow * CDIM;
#pragma unroll
    for (int j = 0; j < 6; ++j) {
        int c = lane + 32 * j;
        o[c] = (_Float16)((v[j] - mu) * inv * g[c] + b[c]);
    }
}

// ---------------------------------------------------------------------------
// Generic WMMA GEMM: Y[M,NC] = A[M,K] @ W[NC,K]^T + bias.
// blockDim 128 (4 wave32); tile 64x64; K stepped by 32 via LDS staging.
// ACT: 0 none, 1 exact GELU.
// OUTMODE: 0 f16 store; 1 f32 residual add; 2 proj: window-reverse + unshift
//          scatter with f32 residual add.
// ---------------------------------------------------------------------------
template <int ACT, int OUTMODE>
__global__ void k_gemm_wmma(const _Float16* __restrict__ A,
                            const _Float16* __restrict__ Bw,
                            const float*    __restrict__ bias,
                            _Float16*       __restrict__ outH,
                            const float*    __restrict__ res,
                            float*          __restrict__ outF,
                            int K, int NC, int shift) {
    __shared__ _Float16 As[64][40];
    __shared__ _Float16 Bs[64][40];

    const int tid  = threadIdx.x;
    const int lane = tid & 31;
    const int wid  = tid >> 5;
    const int hl   = lane >> 4;
    const int l15  = lane & 15;
    const int rowBase = blockIdx.y * 64;
    const int colBase = blockIdx.x * 64;

    v8f acc[4];
#pragma unroll
    for (int ni = 0; ni < 4; ++ni) acc[ni] = {};

    for (int k0 = 0; k0 < K; k0 += 32) {
        for (int i = tid; i < 1024; i += 128) {
            int r = i >> 4, c = (i & 15) << 1;
            union { uint32_t u; _Float16 h[2]; } cv;
            cv.u = *reinterpret_cast<const uint32_t*>(
                       &A[(size_t)(rowBase + r) * K + k0 + c]);
            As[r][c] = cv.h[0]; As[r][c + 1] = cv.h[1];
            if (k0 + 32 < K)
                __builtin_prefetch(&A[(size_t)(rowBase + r) * K + k0 + 32 + c], 0, 1);
        }
        for (int i = tid; i < 1024; i += 128) {
            int n = i >> 4, c = (i & 15) << 1;
            union { uint32_t u; _Float16 h[2]; } cv;
            cv.u = *reinterpret_cast<const uint32_t*>(
                       &Bw[(size_t)(colBase + n) * K + k0 + c]);
            Bs[n][c] = cv.h[0]; Bs[n][c + 1] = cv.h[1];
        }
        __syncthreads();

        v16h a;
#pragma unroll
        for (int e = 0; e < 16; ++e) a[e] = As[16 * wid + l15][kmapA(e, hl)];
#pragma unroll
        for (int ni = 0; ni < 4; ++ni) {
            v16h bfr;
#pragma unroll
            for (int e = 0; e < 16; ++e) bfr[e] = Bs[16 * ni + l15][kmapB(e, hl)];
            acc[ni] = wmma_f16(a, bfr, acc[ni]);
        }
        __syncthreads();
    }

    // epilogue
#pragma unroll
    for (int ni = 0; ni < 4; ++ni) {
        int col = colBase + 16 * ni + l15;
        float bv = bias[col];
#pragma unroll
        for (int r = 0; r < 8; ++r) {
            int row = rowBase + 16 * wid + r + 8 * hl;
            float val = acc[ni][r] + bv;
            if constexpr (ACT == 1)  // exact GELU (erf)
                val = 0.5f * val * (1.0f + erff(val * 0.70710678118654752f));
            if constexpr (OUTMODE == 0) {
                outH[(size_t)row * NC + col] = (_Float16)val;
            } else if constexpr (OUTMODE == 1) {
                size_t idx = (size_t)row * NC + col;
                outF[idx] = res[idx] + val;
            } else {  // proj: window-reverse + un-roll scatter + residual
                int bb  = row >> 12;
                int rem = row & 4095;
                int win = rem >> 6, n = rem & 63;
                int gh = ((win >> 3) << 3) + (n >> 3);
                int gw = ((win & 7) << 3) + (n & 7);
                int oh = (gh + shift) & 63;
                int ow = (gw + shift) & 63;
                size_t idx = ((size_t)bb * 4096 + oh * 64 + ow) * CDIM + col;
                outF[idx] = res[idx] + val;
            }
        }
    }
}

// ---------------------------------------------------------------------------
// Windowed attention, one (window, head) per 128-thread block (4 wave32).
// Q/K/V staged LDS via Tensor Data Mover (hardware-padded rows); scores via
// 4x4 K=32 WMMA tiles + precomputed bias/mask; P@V via 4x2 chained WMMAs.
// Dynamic LDS layout (byte offsets):
//   Qs [64][40] f16 @ 0      Ks @ 5120      Vs @ 10240
//   Sf [64][65] f32 @ 15360  Pw [64][72] f16 @ 32000   (total 41216 B)
// ---------------------------------------------------------------------------
#define QS_OFF 0u
#define KS_OFF 5120u
#define VS_OFF 10240u
#define SF_OFF 15360u
#define PW_OFF 32000u
#define ATTN_LDS_BYTES 41216u

__global__ void k_attention(const _Float16* __restrict__ qkv,
                            const float*    __restrict__ bm,
                            _Float16*       __restrict__ o) {
    extern __shared__ char smem[];
    _Float16 (*Qs)[40] = (_Float16(*)[40])(smem + QS_OFF);
    _Float16 (*Ks)[40] = (_Float16(*)[40])(smem + KS_OFF);
    _Float16 (*Vs)[40] = (_Float16(*)[40])(smem + VS_OFF);
    float    (*Sf)[65] = (float   (*)[65])(smem + SF_OFF);
    _Float16 (*Pw)[72] = (_Float16(*)[72])(smem + PW_OFF);

    const int blk  = blockIdx.x;
    const int win  = blk / NHEADS;       // global window id (b*NWIN + w)
    const int head = blk % NHEADS;
    const int tid  = threadIdx.x;
    const int lane = tid & 31;
    const int wid  = tid >> 5;
    const int hl   = lane >> 4;
    const int l15  = lane & 15;

    // ---- TDM: DMA Q/K/V 64x32 tiles into padded LDS; wave 0 issues ----
    const _Float16* base = qkv + (size_t)win * NTOK * QKVDIM + head * HDIM;
    if (wid == 0) {
        tdm_load_qkv_tile(QS_OFF, base);
        tdm_load_qkv_tile(KS_OFF, base + CDIM);
        tdm_load_qkv_tile(VS_OFF, base + 2 * CDIM);
        __builtin_amdgcn_s_wait_tensorcnt(0);
    }
    __syncthreads();

    // ---- S = scale*(Q K^T) + bias + mask ----
    const int wwin = win & 63;
    const float* bmtile = bm + ((size_t)head * NWIN + wwin) * (NTOK * NTOK);
    const float scale = 0.17677669529663687f;  // 32^-0.5 (folded out of Q)

    v16h a;
#pragma unroll
    for (int e = 0; e < 16; ++e) a[e] = Qs[16 * wid + l15][kmapA(e, hl)];

#pragma unroll
    for (int ni = 0; ni < 4; ++ni) {
        v16h bfr;
#pragma unroll
        for (int e = 0; e < 16; ++e) bfr[e] = Ks[16 * ni + l15][kmapB(e, hl)];
        v8f acc = {};
        acc = wmma_f16(a, bfr, acc);
#pragma unroll
        for (int r = 0; r < 8; ++r) {
            int i_ = 16 * wid + r + 8 * hl;   // query token in window
            int j_ = 16 * ni + l15;           // key token in window
            Sf[i_][j_] = acc[r] * scale + bmtile[i_ * NTOK + j_];
        }
    }
    __syncthreads();

    // ---- row softmax (64 rows, one thread each) ----
    if (tid < NTOK) {
        float mx = -1e30f;
        for (int j = 0; j < NTOK; ++j) mx = fmaxf(mx, Sf[tid][j]);
        float sm = 0.f;
        for (int j = 0; j < NTOK; ++j) {
            float e = __expf(Sf[tid][j] - mx);
            Sf[tid][j] = e;
            sm += e;
        }
        float inv = 1.0f / sm;
        for (int j = 0; j < NTOK; ++j) Pw[tid][j] = (_Float16)(Sf[tid][j] * inv);
    }
    __syncthreads();

    // ---- O = P V ----
#pragma unroll
    for (int vi = 0; vi < 2; ++vi) {
        v8f acc = {};
#pragma unroll
        for (int kc = 0; kc < 2; ++kc) {
            v16h ap, bp;
#pragma unroll
            for (int e = 0; e < 16; ++e)
                ap[e] = Pw[16 * wid + l15][32 * kc + kmapA(e, hl)];
#pragma unroll
            for (int e = 0; e < 16; ++e)
                bp[e] = Vs[32 * kc + kmapB(e, hl)][16 * vi + l15];
            acc = wmma_f16(ap, bp, acc);
        }
#pragma unroll
        for (int r = 0; r < 8; ++r) {
            int n = 16 * wid + r + 8 * hl;
            int d = 16 * vi + l15;
            o[((size_t)win * NTOK + n) * CDIM + head * HDIM + d] = (_Float16)acc[r];
        }
    }
}

// ---------------------------------------------------------------------------
// Host orchestration
// ---------------------------------------------------------------------------
static void run_gemm(int act, int outmode,
                     const _Float16* A, const _Float16* Bw, const float* bias,
                     _Float16* outH, const float* res, float* outF,
                     int M, int K, int NC, int shift, hipStream_t s) {
    dim3 g(NC / 64, M / 64), b(128);
    if (act == 1 && outmode == 0)
        k_gemm_wmma<1, 0><<<g, b, 0, s>>>(A, Bw, bias, outH, res, outF, K, NC, shift);
    else if (act == 0 && outmode == 0)
        k_gemm_wmma<0, 0><<<g, b, 0, s>>>(A, Bw, bias, outH, res, outF, K, NC, shift);
    else if (outmode == 1)
        k_gemm_wmma<0, 1><<<g, b, 0, s>>>(A, Bw, bias, outH, res, outF, K, NC, shift);
    else
        k_gemm_wmma<0, 2><<<g, b, 0, s>>>(A, Bw, bias, outH, res, outF, K, NC, shift);
}

extern "C" void kernel_launch(void* const* d_in, const int* in_sizes, int n_in,
                              void* d_out, int out_size, void* d_ws, size_t ws_size,
                              hipStream_t stream) {
    const float* x0      = (const float*)d_in[0];
    const float* n1g     = (const float*)d_in[1];
    const float* n1b     = (const float*)d_in[2];
    const float* qkv_w   = (const float*)d_in[3];
    const float* qkv_b   = (const float*)d_in[4];
    const float* btbl    = (const float*)d_in[5];
    const float* proj_w  = (const float*)d_in[6];
    const float* proj_b  = (const float*)d_in[7];
    const float* n2g     = (const float*)d_in[8];
    const float* n2b     = (const float*)d_in[9];
    const float* fc1_w   = (const float*)d_in[10];
    const float* fc1_b   = (const float*)d_in[11];
    const float* fc2_w   = (const float*)d_in[12];
    const float* fc2_b   = (const float*)d_in[13];
    float* xout = (float*)d_out;

    // workspace carve-up
    size_t off = 0;
    auto carve = [&](size_t bytes) {
        void* p = (char*)d_ws + off;
        off += (bytes + 255) & ~(size_t)255;
        return p;
    };
    _Float16* wqkv = (_Float16*)carve(2 * QKVDIM * CDIM * sizeof(_Float16));
    _Float16* wprj = (_Float16*)carve(2 * CDIM * CDIM * sizeof(_Float16));
    _Float16* wfc1 = (_Float16*)carve(2 * FFDIM * CDIM * sizeof(_Float16));
    _Float16* wfc2 = (_Float16*)carve(2 * CDIM * FFDIM * sizeof(_Float16));
    _Float16* winX = (_Float16*)carve((size_t)TOKENS * CDIM * sizeof(_Float16));
    _Float16* qkvB = (_Float16*)carve((size_t)TOKENS * QKVDIM * sizeof(_Float16));
    _Float16* attO = (_Float16*)carve((size_t)TOKENS * CDIM * sizeof(_Float16));
    _Float16* ln2B = (_Float16*)carve((size_t)TOKENS * CDIM * sizeof(_Float16));
    _Float16* hB   = (_Float16*)carve((size_t)TOKENS * FFDIM * sizeof(_Float16));
    float*    xmid = (float*)carve((size_t)TOKENS * CDIM * sizeof(float));
    float*    bmB  = (float*)carve((size_t)NHEADS * NWIN * NTOK * NTOK * sizeof(float));

    auto conv = [&](const float* s, _Float16* d, int n) {
        k_convert_f16<<<(n + 255) / 256, 256, 0, stream>>>(s, d, n);
    };
    conv(qkv_w,  wqkv, 2 * QKVDIM * CDIM);
    conv(proj_w, wprj, 2 * CDIM * CDIM);
    conv(fc1_w,  wfc1, 2 * FFDIM * CDIM);
    conv(fc2_w,  wfc2, 2 * CDIM * FFDIM);

    const float* xcur = x0;
    for (int dep = 0; dep < 2; ++dep) {
        int shift = (dep == 0) ? 0 : 4;

        // bias + mask tensor for this depth
        k_biasmask<<<NHEADS * NWIN, 256, 0, stream>>>(
            btbl + (size_t)dep * TBLSZ * NHEADS, bmB, shift);

        // LN1 + shift + window partition -> f16
        k_layernorm<<<TOKENS / 8, 256, 0, stream>>>(
            xcur, n1g + dep * CDIM, n1b + dep * CDIM, winX, shift, 1);

        // qkv = winX @ Wqkv^T + b
        run_gemm(0, 0, winX, wqkv + (size_t)dep * QKVDIM * CDIM,
                 qkv_b + dep * QKVDIM, qkvB, nullptr, nullptr,
                 TOKENS, CDIM, QKVDIM, 0, stream);

        // attention (TDM-staged QKV, precomputed bias/mask, softmax, PV)
        k_attention<<<(TOKENS / NTOK) * NHEADS, 128, ATTN_LDS_BYTES, stream>>>(
            qkvB, bmB, attO);

        // proj + window-reverse + unshift + residual -> xmid (f32)
        run_gemm(0, 2, attO, wprj + (size_t)dep * CDIM * CDIM,
                 proj_b + dep * CDIM, nullptr, xcur, xmid,
                 TOKENS, CDIM, CDIM, shift, stream);

        // LN2 -> f16
        k_layernorm<<<TOKENS / 8, 256, 0, stream>>>(
            xmid, n2g + dep * CDIM, n2b + dep * CDIM, ln2B, 0, 0);

        // fc1 + exact GELU
        run_gemm(1, 0, ln2B, wfc1 + (size_t)dep * FFDIM * CDIM,
                 fc1_b + dep * FFDIM, hB, nullptr, nullptr,
                 TOKENS, CDIM, FFDIM, 0, stream);

        // fc2 + residual -> xout
        run_gemm(0, 1, hB, wfc2 + (size_t)dep * CDIM * FFDIM,
                 fc2_b + dep * CDIM, nullptr, xmid, xout,
                 TOKENS, FFDIM, CDIM, 0, stream);

        xcur = xout;  // next block reads this
    }
    (void)in_sizes; (void)n_in; (void)out_size; (void)ws_size;
}
